// Aligner_66709432042261
// MI455X (gfx1250) — compile-verified, matrix operand order checked by reading, not run on hardware
//
#include <hip/hip_runtime.h>
#include <hip/hip_bf16.h>

// Problem constants (match reference)
#define NB 16
#define ND 512
#define NT 2048
#define NL 512
#define NSTRIDE 4
#define SIGMA_SQ 5.0f

typedef __attribute__((ext_vector_type(16))) __bf16 v16bf;
typedef __attribute__((ext_vector_type(8)))  float  v8f;

// ---------------------------------------------------------------------------
// Kernel 1: score[b,t] = (t < len) ? exp(sum_d x[b,d,t]*w[d]) : 0
// ---------------------------------------------------------------------------
__global__ __launch_bounds__(256) void score_kernel(const float* __restrict__ x,
                                                    const float* __restrict__ w,
                                                    const int*   __restrict__ xlen,
                                                    float*       __restrict__ score) {
    const int b = blockIdx.y;
    const int t = blockIdx.x * 256 + threadIdx.x;
    __shared__ float wsm[ND];
    for (int d = threadIdx.x; d < ND; d += 256) wsm[d] = w[d];
    __syncthreads();
    const float* xp = x + (size_t)b * ND * NT + t;
    float acc = 0.f;
#pragma unroll 8
    for (int d = 0; d < ND; ++d) acc += wsm[d] * xp[(size_t)d * NT];
    const int len = xlen[b];
    score[b * NT + t] = (t < len) ? __expf(acc) : 0.f;
}

// ---------------------------------------------------------------------------
// Kernel 2: per-batch inclusive scan -> norm[b,t]; loss partial; z_mask; z_len
// One block of 1024 threads per batch; each thread owns 2 consecutive t.
// ---------------------------------------------------------------------------
__global__ __launch_bounds__(1024) void scan_kernel(const float* __restrict__ score,
                                                    const int*   __restrict__ xlen,
                                                    float* __restrict__ norm,
                                                    float* __restrict__ lossb,
                                                    float* __restrict__ out_zmask,
                                                    float* __restrict__ out_zlen) {
    const int b   = blockIdx.x;
    const int tid = threadIdx.x;
    __shared__ float sums[1024];
    const float* sc = score + b * NT;
    const float a0 = sc[2 * tid];
    const float a1 = sc[2 * tid + 1];
    const float s  = a0 + a1;
    sums[tid] = s;
    __syncthreads();
    // Hillis-Steele inclusive scan over the 1024 per-thread sums
    for (int off = 1; off < 1024; off <<= 1) {
        float v = (tid >= off) ? sums[tid - off] : 0.f;
        __syncthreads();
        sums[tid] += v;
        __syncthreads();
    }
    const float total = sums[1023];
    const float excl  = sums[tid] - s;

    const int   len   = xlen[b];
    const int   zlen  = (len + NSTRIDE - 1) / NSTRIDE;
    const float cum0  = sc[0];
    const float scale = ((float)zlen - 1.f) / (total - cum0);

    const float c0 = excl + a0;
    const float c1 = excl + a0 + a1;
    norm[b * NT + 2 * tid]     = (c0 - cum0) * scale;
    norm[b * NT + 2 * tid + 1] = (c1 - cum0) * scale;

    // diff[t] = norm[t]-norm[t-1] = score[t]*scale  (t >= 1)
    float contrib = 0.f;
    const int t0 = 2 * tid, t1 = 2 * tid + 1;
    if (t0 >= 1 && t0 < len) contrib += fmaxf(a0 * scale - 1.f, 0.f);
    if (t1 >= 1 && t1 < len) contrib += fmaxf(a1 * scale - 1.f, 0.f);
    __syncthreads();
    sums[tid] = contrib;
    __syncthreads();
    for (int s2 = 512; s2 > 0; s2 >>= 1) {
        if (tid < s2) sums[tid] += sums[tid + s2];
        __syncthreads();
    }
    if (tid == 0) {
        lossb[b]    = sums[0] / ((float)len - 1.f);
        out_zlen[b] = (float)zlen;
    }
    if (tid < NL) out_zmask[b * NL + tid] = (tid * NSTRIDE < len) ? 1.f : 0.f;
}

// ---------------------------------------------------------------------------
// Kernel 3: alignment[b,l,t] = softmax_t(-5*(l - norm[b,t])^2, mask) * zmask
// One 256-thread block per (b,l); exp() results staged in LDS.
// ---------------------------------------------------------------------------
__global__ __launch_bounds__(256) void softmax_kernel(const float* __restrict__ norm,
                                                      const int*   __restrict__ xlen,
                                                      float*       __restrict__ align) {
    const int b = blockIdx.y, l = blockIdx.x, tid = threadIdx.x;
    __shared__ float p[NT];
    __shared__ float red[256];
    const float* nb = norm + b * NT;
    const int len = xlen[b];
    const float lf = (float)l;

    float m = -3.4e38f;
    for (int t = tid; t < NT; t += 256) {
        const float df = lf - nb[t];
        const float d  = -SIGMA_SQ * df * df;
        if (t < len) m = fmaxf(m, d);
    }
    red[tid] = m;
    __syncthreads();
    for (int s = 128; s > 0; s >>= 1) {
        if (tid < s) red[tid] = fmaxf(red[tid], red[tid + s]);
        __syncthreads();
    }
    m = red[0];
    __syncthreads();

    float sum = 0.f;
    for (int t = tid; t < NT; t += 256) {
        const float df = lf - nb[t];
        const float d  = -SIGMA_SQ * df * df;
        const float e  = (t < len) ? __expf(d - m) : 0.f;
        p[t] = e;
        sum += e;
    }
    red[tid] = sum;
    __syncthreads();
    for (int s = 128; s > 0; s >>= 1) {
        if (tid < s) red[tid] += red[tid + s];
        __syncthreads();
    }
    sum = red[0];

    const int   zlen = (len + NSTRIDE - 1) / NSTRIDE;
    const float inv  = (l < zlen) ? 1.f / sum : 0.f;  // folds z_mask into output
    float* ap = align + ((size_t)(b * NL + l)) * NT;
    for (int t = tid; t < NT; t += 256) ap[t] = p[t] * inv;
}

// ---------------------------------------------------------------------------
// Kernel 4: score_loss = mean_b lossb[b]
// ---------------------------------------------------------------------------
__global__ void finalize_kernel(const float* __restrict__ lossb,
                                float* __restrict__ out_loss) {
    if (threadIdx.x == 0) {
        float s = 0.f;
        for (int i = 0; i < NB; ++i) s += lossb[i];
        out_loss[0] = s / (float)NB;
    }
}

// ---------------------------------------------------------------------------
// Kernel 5: z[b,d,l] = sum_t x[b,d,t] * align[b,l,t]
// Batched GEMM M=512, N=512, K=2048 via v_wmma_f32_16x16x32_bf16.
// Wave tile 32(d) x 64(l): 2 A frags, 4 B frags, 8 accumulators.
// Block = 8 waves -> 128x128 tile. f32 -> bf16 conversion in registers.
// ---------------------------------------------------------------------------
__device__ inline v16bf load_cvt_frag(const float* __restrict__ p) {
    // p points at the lane's first K element (k0 + hi*8).
    // frag elements [0..7]  = K (k0 + hi*8 + 0..7)
    // frag elements [8..15] = K (k0 + 16 + hi*8 + 0..7)
    const float4 f0 = *(const float4*)(p);
    const float4 f1 = *(const float4*)(p + 4);
    const float4 f2 = *(const float4*)(p + 16);
    const float4 f3 = *(const float4*)(p + 20);
    v16bf r;
    r[0]  = (__bf16)f0.x; r[1]  = (__bf16)f0.y; r[2]  = (__bf16)f0.z; r[3]  = (__bf16)f0.w;
    r[4]  = (__bf16)f1.x; r[5]  = (__bf16)f1.y; r[6]  = (__bf16)f1.z; r[7]  = (__bf16)f1.w;
    r[8]  = (__bf16)f2.x; r[9]  = (__bf16)f2.y; r[10] = (__bf16)f2.z; r[11] = (__bf16)f2.w;
    r[12] = (__bf16)f3.x; r[13] = (__bf16)f3.y; r[14] = (__bf16)f3.z; r[15] = (__bf16)f3.w;
    return r;
}

__global__ __launch_bounds__(256) void zgemm_kernel(const float* __restrict__ x,
                                                    const float* __restrict__ align,
                                                    float* __restrict__ z) {
    const int b    = blockIdx.z;
    const int dblk = blockIdx.x * 128;
    const int lblk = blockIdx.y * 128;
    const int tid  = threadIdx.x;
    const int wave = tid >> 5;
    const int lane = tid & 31;
    const int laneM = lane & 15;
    const int hi    = lane >> 4;
    const int wx = wave >> 1;           // 0..3 : d dimension
    const int wy = wave & 1;            // 0..1 : l dimension
    const int d0 = dblk + wx * 32;
    const int l0 = lblk + wy * 64;

    v8f c[2][4];
#pragma unroll
    for (int i = 0; i < 2; ++i)
#pragma unroll
        for (int j = 0; j < 4; ++j) c[i][j] = (v8f)(0.f);

    const float* xrow[2];
#pragma unroll
    for (int i = 0; i < 2; ++i)
        xrow[i] = x + ((size_t)(b * ND + d0 + i * 16 + laneM)) * NT + hi * 8;
    const float* arow[4];
#pragma unroll
    for (int j = 0; j < 4; ++j)
        arow[j] = align + ((size_t)(b * NL + l0 + j * 16 + laneM)) * NT + hi * 8;

#pragma unroll 2
    for (int k0 = 0; k0 < NT; k0 += 32) {
        v16bf af[2], bfr[4];
#pragma unroll
        for (int i = 0; i < 2; ++i) af[i]  = load_cvt_frag(xrow[i] + k0);
#pragma unroll
        for (int j = 0; j < 4; ++j) bfr[j] = load_cvt_frag(arow[j] + k0);
#pragma unroll
        for (int i = 0; i < 2; ++i)
#pragma unroll
            for (int j = 0; j < 4; ++j)
                c[i][j] = __builtin_amdgcn_wmma_f32_16x16x32_bf16(
                    false, af[i], false, bfr[j], (short)0, c[i][j], false, false);
    }

    // C layout: VGPR r holds M = r + hi*8 (row of Z), lane laneM gives N (col).
#pragma unroll
    for (int i = 0; i < 2; ++i)
#pragma unroll
        for (int j = 0; j < 4; ++j) {
            const int dd = d0 + i * 16 + hi * 8;
            const int ll = l0 + j * 16 + laneM;
            float* zp = z + ((size_t)(b * ND + dd)) * NL + ll;
#pragma unroll
            for (int r = 0; r < 8; ++r) zp[(size_t)r * NL] = c[i][j][r];
        }
}

// ---------------------------------------------------------------------------
// Host-side launcher
// ---------------------------------------------------------------------------
extern "C" void kernel_launch(void* const* d_in, const int* in_sizes, int n_in,
                              void* d_out, int out_size, void* d_ws, size_t ws_size,
                              hipStream_t stream) {
    const float* x    = (const float*)d_in[0];
    const float* w    = (const float*)d_in[1];
    // d_in[2] is x_mask (bool); mask is re-derived from x_lengths instead.
    const int*   xlen = (const int*)d_in[3];

    // Output tuple, concatenated flat as f32:
    // z (B*D*L), z_mask (B*L), z_lengths (B), alignment (B*L*T), score_loss (1)
    float* out      = (float*)d_out;
    float* z        = out;
    float* zmask    = z + (size_t)NB * ND * NL;
    float* zlen_out = zmask + (size_t)NB * NL;
    float* align    = zlen_out + NB;
    float* loss     = align + (size_t)NB * NL * NT;

    // Workspace: score (B*T), norm (B*T), per-batch loss (B)  -> ~256 KB
    float* score = (float*)d_ws;
    float* norm  = score + (size_t)NB * NT;
    float* lossb = norm + (size_t)NB * NT;

    score_kernel   <<<dim3(NT / 256, NB), 256, 0, stream>>>(x, w, xlen, score);
    scan_kernel    <<<dim3(NB), 1024, 0, stream>>>(score, xlen, norm, lossb, zmask, zlen_out);
    softmax_kernel <<<dim3(NL, NB), 256, 0, stream>>>(norm, xlen, align);
    finalize_kernel<<<1, 32, 0, stream>>>(lossb, loss);
    zgemm_kernel   <<<dim3(ND / 128, NL / 128, NB), 256, 0, stream>>>(x, align, z);
}
